// HysteresisRouter_70523363000766
// MI455X (gfx1250) — compile-verified, matrix operand order checked by reading, not run on hardware
//
#include <hip/hip_runtime.h>
#include <hip/hip_bf16.h>
#include <stddef.h>

// ---------------------------------------------------------------------------
// HysteresisRouter on MI455X (gfx1250, wave32, WMMA)
//   N=16384 tokens, D=2048, E=64 experts, TOP_K=2, 10 Sinkhorn iters
// ---------------------------------------------------------------------------

#define N_TOK   16384
#define DDIM    2048
#define NEXP    64
#define LAMBDA_TAX 0.04f
#define SK_EPS  1e-12f

typedef __attribute__((ext_vector_type(16))) _Float16 v16h;
typedef __attribute__((ext_vector_type(8)))  _Float16 v8h;
typedef __attribute__((ext_vector_type(8)))  float    v8f;

union Frag16 { v16h v; v8h h[2]; };

// ---------------------------------------------------------------------------
// 0) zero scratch: C[64*64] and 10 column-sum slots (10*64)
// ---------------------------------------------------------------------------
__global__ void k_zero(float* __restrict__ C, float* __restrict__ cs) {
    int t = blockIdx.x * 256 + threadIdx.x;
    if (t < NEXP * NEXP) C[t]  = 0.0f;
    if (t < 10 * NEXP)   cs[t] = 0.0f;
}

// ---------------------------------------------------------------------------
// 1) logits = x@W + b  (WMMA f16 -> f32), center, softmax.
//    Block: 256 threads (8 waves). Block tile: 128 rows x 64 experts.
//    Wave w owns rows [w*16, w*16+16) x all 64 cols (4 accumulators).
// ---------------------------------------------------------------------------
__global__ __launch_bounds__(256)
void k_gemm_softmax(const float* __restrict__ x, const float* __restrict__ W,
                    const float* __restrict__ b,
                    float* __restrict__ L, float* __restrict__ M) {
    // padded LDS tiles (stride 40 halfs = 80 B/row, keeps ds_load_b128 conflict-free)
    __shared__ __align__(16) _Float16 As[128 * 40];  // x tile,   [row][k]
    __shared__ __align__(16) _Float16 Bs[64 * 40];   // W tile^T, [n][k]

    const int t       = threadIdx.x;
    const int wave    = t >> 5;
    const int lane    = t & 31;
    const int lr      = lane & 15;
    const int kb      = (lane < 16) ? 0 : 8;   // ISA 16-bit A/B lane K-offset
    const int rowBase = blockIdx.x * 128;

    v8f acc[4] = {};

    for (int kk = 0; kk < DDIM; kk += 32) {
        // --- A tile: 128 rows x 32 k, fp32 -> f16, 2 threads per row ---
        {
            const int r  = t >> 1;
            const int c0 = (t & 1) * 16;
            const float4* src =
                (const float4*)(x + (size_t)(rowBase + r) * DDIM + kk + c0);
            _Float16* dst = &As[r * 40 + c0];
#pragma unroll
            for (int j = 0; j < 4; ++j) {
                float4 v = src[j];
                dst[j * 4 + 0] = (_Float16)v.x;
                dst[j * 4 + 1] = (_Float16)v.y;
                dst[j * 4 + 2] = (_Float16)v.z;
                dst[j * 4 + 3] = (_Float16)v.w;
            }
        }
        // --- B tile: W[kk..kk+31][0..63] -> Bs[n][k] (transposed) ---
        {
            const int kl = t >> 3;          // 0..31
            const int n0 = (t & 7) * 8;     // 0..56
            const float* src = W + (size_t)(kk + kl) * NEXP + n0;
#pragma unroll
            for (int j = 0; j < 8; ++j)
                Bs[(n0 + j) * 40 + kl] = (_Float16)src[j];
        }
        __syncthreads();

        // --- fragment gathers: two 16B chunks per lane (K kb..kb+7, kb+16..kb+23)
        Frag16 af, bf[4];
        const _Float16* ap = &As[(wave * 16 + lr) * 40 + kb];
        af.h[0] = *(const v8h*)(ap);
        af.h[1] = *(const v8h*)(ap + 16);
#pragma unroll
        for (int nt = 0; nt < 4; ++nt) {
            const _Float16* bp = &Bs[(nt * 16 + lr) * 40 + kb];
            bf[nt].h[0] = *(const v8h*)(bp);
            bf[nt].h[1] = *(const v8h*)(bp + 16);
        }
#pragma unroll
        for (int nt = 0; nt < 4; ++nt) {
            acc[nt] = __builtin_amdgcn_wmma_f32_16x16x32_f16(
                false, af.v, false, bf[nt].v, (short)0, acc[nt], false, false);
        }
        __syncthreads();
    }

    // --- epilogue: +bias, center, softmax. C-layout: vgpr i holds row i+8*(lane/16),
    //     col nt*16 + (lane&15). Row lives in one 16-lane half -> xor masks 1..8.
    const int half16 = lane >> 4;
    float bias[4];
#pragma unroll
    for (int nt = 0; nt < 4; ++nt) bias[nt] = b[nt * 16 + lr];

#pragma unroll
    for (int i = 0; i < 8; ++i) {
        float v0 = acc[0][i] + bias[0];
        float v1 = acc[1][i] + bias[1];
        float v2 = acc[2][i] + bias[2];
        float v3 = acc[3][i] + bias[3];

        float s = v0 + v1 + v2 + v3;
#pragma unroll
        for (int m = 1; m < 16; m <<= 1) s += __shfl_xor(s, m, 32);
        const float mean = s * (1.0f / 64.0f);
        v0 -= mean; v1 -= mean; v2 -= mean; v3 -= mean;

        float mx = fmaxf(fmaxf(v0, v1), fmaxf(v2, v3));
#pragma unroll
        for (int m = 1; m < 16; m <<= 1) mx = fmaxf(mx, __shfl_xor(mx, m, 32));
        float e0 = __expf(v0 - mx), e1 = __expf(v1 - mx);
        float e2 = __expf(v2 - mx), e3 = __expf(v3 - mx);
        float es = e0 + e1 + e2 + e3;
#pragma unroll
        for (int m = 1; m < 16; m <<= 1) es += __shfl_xor(es, m, 32);
        const float inv = 1.0f / es;

        const int row = rowBase + wave * 16 + i + half16 * 8;
        float* Lr = L + (size_t)row * NEXP + lr;
        float* Mr = M + (size_t)row * NEXP + lr;
        Lr[0] = v0; Lr[16] = v1; Lr[32] = v2; Lr[48] = v3;
        Mr[0] = e0 * inv; Mr[16] = e1 * inv; Mr[32] = e2 * inv; Mr[48] = e3 * inv;
    }
}

// ---------------------------------------------------------------------------
// 2) C = M^T @ M  (64x64). 64 blocks x 256 rows; LDS tiles + global atomics.
// ---------------------------------------------------------------------------
__global__ __launch_bounds__(256)
void k_mtm(const float* __restrict__ M, float* __restrict__ C) {
    __shared__ float Ms[64][65];
    const int t   = threadIdx.x;
    const int e1  = t & 63;
    const int e2b = (t >> 6) * 16;
    float acc[16];
#pragma unroll
    for (int j = 0; j < 16; ++j) acc[j] = 0.0f;

    const int rows0 = blockIdx.x * 256;
    for (int tile = 0; tile < 4; ++tile) {
        const int r0 = rows0 + tile * 64;
        for (int idx = t; idx < 64 * 64; idx += 256) {
            int r = idx >> 6, c = idx & 63;
            Ms[r][c] = M[(size_t)(r0 + r) * NEXP + c];
        }
        __syncthreads();
        for (int r = 0; r < 64; ++r) {
            float a = Ms[r][e1];
#pragma unroll
            for (int j = 0; j < 16; ++j) acc[j] += a * Ms[r][e2b + j];
        }
        __syncthreads();
    }
#pragma unroll
    for (int j = 0; j < 16; ++j) atomicAdd(&C[e1 * 64 + e2b + j], acc[j]);
}

// ---------------------------------------------------------------------------
// 3) Tax-gradient step + re-softmax (in-place M), fused colsum[0] accumulate.
//    256 blocks x 64 rows. C cached in LDS with diag zeroed at load.
// ---------------------------------------------------------------------------
__global__ __launch_bounds__(256)
void k_taxgrad(float* __restrict__ M, const float* __restrict__ L,
               const float* __restrict__ C, float* __restrict__ cs0) {
    __shared__ float Cs[64][65];
    __shared__ float Ms[64][65];
    __shared__ float Gs[64][65];
    const int t  = threadIdx.x;
    const int r0 = blockIdx.x * 64;

    for (int idx = t; idx < 64 * 64; idx += 256) {
        int i = idx >> 6, j = idx & 63;
        Cs[i][j] = (i == j) ? 0.0f : C[idx];
        Ms[i][j] = M[(size_t)(r0 + i) * NEXP + j];
    }
    __syncthreads();

    // G = 4 * Ms @ Cs : 16 entries per thread
    {
        const int e  = t & 63;
        const int rb = (t >> 6) * 16;
        for (int rr = 0; rr < 16; ++rr) {
            const int r = rb + rr;
            float s = 0.0f;
            for (int k = 0; k < 64; ++k) s += Ms[r][k] * Cs[k][e];
            Gs[r][e] = 4.0f * s;
        }
    }
    __syncthreads();

    // exact_grad, logits update, softmax: one thread per row (reuses Gs as scratch)
    if (t < 64) {
        const int r = t;
        float dot = 0.0f;
        for (int e = 0; e < 64; ++e) dot += Ms[r][e] * Gs[r][e];
        const float* Lr = L + (size_t)(r0 + r) * NEXP;
        float mx = -1e30f;
        for (int e = 0; e < 64; ++e) {
            float ex = Ms[r][e] * (Gs[r][e] - dot);
            float v  = Lr[e] - LAMBDA_TAX * ex;
            Gs[r][e] = v;
            mx = fmaxf(mx, v);
        }
        float s = 0.0f;
        for (int e = 0; e < 64; ++e) {
            float q = __expf(Gs[r][e] - mx);
            Gs[r][e] = q;
            s += q;
        }
        const float inv = 1.0f / s;
        float* Mr = M + (size_t)(r0 + r) * NEXP;
        for (int e = 0; e < 64; ++e) {
            float q = Gs[r][e] * inv;
            Mr[e]    = q;
            Ms[r][e] = q;   // keep for colsum phase
        }
    }
    __syncthreads();

    if (t < 64) {
        float s = 0.0f;
        for (int r = 0; r < 64; ++r) s += Ms[r][t];
        atomicAdd(&cs0[t], s);
    }
}

// ---------------------------------------------------------------------------
// 4) One Sinkhorn iteration (col-divide * scale, row-normalize), in-place,
//    fused with NEXT iteration's column-sum accumulation.
//    256 blocks x (8 waves x 8 rows). Lane l owns experts l and l+32.
// ---------------------------------------------------------------------------
__global__ __launch_bounds__(256)
void k_sinkhorn(float* __restrict__ M, const float* __restrict__ csIn,
                float* __restrict__ csOut) {
    __shared__ float csl[64];
    __shared__ float accl[64];
    const int t = threadIdx.x;
    if (t < 64) { csl[t] = csIn[t]; accl[t] = 0.0f; }
    __syncthreads();

    const int wave = t >> 5, lane = t & 31;
    const float scale = (float)NEXP / (float)N_TOK;
    const float inv0 = scale / fmaxf(csl[lane],      SK_EPS);
    const float inv1 = scale / fmaxf(csl[lane + 32], SK_EPS);

    float a0 = 0.0f, a1 = 0.0f;
    const int rbase = blockIdx.x * 64 + wave * 8;
    for (int rr = 0; rr < 8; ++rr) {
        const size_t off = (size_t)(rbase + rr) * NEXP;
        float m0 = M[off + lane]      * inv0;
        float m1 = M[off + lane + 32] * inv1;
        float rs = m0 + m1;
#pragma unroll
        for (int m = 1; m < 32; m <<= 1) rs += __shfl_xor(rs, m, 32);
        const float ri = 1.0f / fmaxf(rs, SK_EPS);
        m0 *= ri; m1 *= ri;
        M[off + lane]      = m0;
        M[off + lane + 32] = m1;
        a0 += m0; a1 += m1;
    }
    atomicAdd(&accl[lane],      a0);   // ds_add_f32 across 8 waves
    atomicAdd(&accl[lane + 32], a1);
    __syncthreads();
    if (t < 64) atomicAdd(&csOut[t], accl[t]);
}

// ---------------------------------------------------------------------------
// 5) Final Sinkhorn iteration fused with top-2 mask; writes d_out directly.
// ---------------------------------------------------------------------------
__global__ __launch_bounds__(256)
void k_final(const float* __restrict__ M, const float* __restrict__ csIn,
             float* __restrict__ outM, float* __restrict__ outMask) {
    __shared__ float csl[64];
    const int t = threadIdx.x;
    if (t < 64) csl[t] = csIn[t];
    __syncthreads();

    const int wave = t >> 5, lane = t & 31;
    const float scale = (float)NEXP / (float)N_TOK;
    const float inv0 = scale / fmaxf(csl[lane],      SK_EPS);
    const float inv1 = scale / fmaxf(csl[lane + 32], SK_EPS);

    const int rbase = blockIdx.x * 64 + wave * 8;
    for (int rr = 0; rr < 8; ++rr) {
        const size_t off = (size_t)(rbase + rr) * NEXP;
        float m0 = M[off + lane]      * inv0;
        float m1 = M[off + lane + 32] * inv1;
        float rs = m0 + m1;
#pragma unroll
        for (int m = 1; m < 32; m <<= 1) rs += __shfl_xor(rs, m, 32);
        const float ri = 1.0f / fmaxf(rs, SK_EPS);
        m0 *= ri; m1 *= ri;
        outM[off + lane]      = m0;
        outM[off + lane + 32] = m1;

        // local sorted pair, then butterfly top-2 merge across the wave
        float v1, v2; int i1, i2;
        if (m0 >= m1) { v1 = m0; i1 = lane;      v2 = m1; i2 = lane + 32; }
        else          { v1 = m1; i1 = lane + 32; v2 = m0; i2 = lane;      }
#pragma unroll
        for (int m = 16; m >= 1; m >>= 1) {
            float ov1 = __shfl_xor(v1, m, 32), ov2 = __shfl_xor(v2, m, 32);
            int   oi1 = __shfl_xor(i1, m, 32), oi2 = __shfl_xor(i2, m, 32);
            float nv1, nv2; int ni1, ni2;
            if (v1 >= ov1) {
                nv1 = v1; ni1 = i1;
                if (v2 >= ov1) { nv2 = v2;  ni2 = i2;  }
                else           { nv2 = ov1; ni2 = oi1; }
            } else {
                nv1 = ov1; ni1 = oi1;
                if (ov2 >= v1) { nv2 = ov2; ni2 = oi2; }
                else           { nv2 = v1;  ni2 = i1;  }
            }
            v1 = nv1; v2 = nv2; i1 = ni1; i2 = ni2;
        }
        outMask[off + lane] =
            (lane == i1 || lane == i2) ? 1.0f : 0.0f;
        outMask[off + lane + 32] =
            ((lane + 32) == i1 || (lane + 32) == i2) ? 1.0f : 0.0f;
    }
}

// ---------------------------------------------------------------------------
// launch
// ---------------------------------------------------------------------------
extern "C" void kernel_launch(void* const* d_in, const int* in_sizes, int n_in,
                              void* d_out, int out_size, void* d_ws, size_t ws_size,
                              hipStream_t stream) {
    const float* x = (const float*)d_in[0];   // [16384, 2048]
    const float* W = (const float*)d_in[1];   // [2048, 64]
    const float* b = (const float*)d_in[2];   // [64]

    float* outM    = (float*)d_out;                       // [16384, 64]
    float* outMask = outM + (size_t)N_TOK * NEXP;         // [16384, 64] (0/1)

    float* Mbuf = (float*)d_ws;                           // 4 MB
    float* Lbuf = Mbuf + (size_t)N_TOK * NEXP;            // 4 MB
    float* Cbuf = Lbuf + (size_t)N_TOK * NEXP;            // 16 KB
    float* cs   = Cbuf + NEXP * NEXP;                     // 10 * 64 floats

    k_zero<<<16, 256, 0, stream>>>(Cbuf, cs);
    k_gemm_softmax<<<N_TOK / 128, 256, 0, stream>>>(x, W, b, Lbuf, Mbuf);
    k_mtm<<<64, 256, 0, stream>>>(Mbuf, Cbuf);
    k_taxgrad<<<N_TOK / 64, 256, 0, stream>>>(Mbuf, Lbuf, Cbuf, cs);
    for (int j = 0; j < 9; ++j)
        k_sinkhorn<<<N_TOK / 64, 256, 0, stream>>>(Mbuf, cs + j * 64,
                                                   cs + (j + 1) * 64);
    k_final<<<N_TOK / 64, 256, 0, stream>>>(Mbuf, cs + 9 * 64, outM, outMask);
}
